// CDN_88261577933286
// MI455X (gfx1250) — matrix operand order repaired and checked by reading, not compiled, not run on hardware
//
#include <hip/hip_runtime.h>
#include <hip/hip_bf16.h>

// ---------------- WMMA bf16 helpers (CDNA5 gfx1250, wave32) ----------------
typedef __attribute__((ext_vector_type(16))) __bf16 v16bf;
typedef __attribute__((ext_vector_type(8)))  float  v8f;

union V16U { v16bf v; unsigned int u[8]; };

__device__ __forceinline__ unsigned short f2bf(float f) {
  unsigned int u = __float_as_uint(f);
  u += 0x7FFFu + ((u >> 16) & 1u);           // round-to-nearest-even
  return (unsigned short)(u >> 16);
}

// A tile 16x32 bf16 from a row-major [*, lda] bf16 buffer at (row 0..15, col k0..k0+31).
// ISA 7.12.2 16-bit A layout: lane m(0..15)/hi half-K; VGPR j holds a K pair.
// The 4 dwords per half are contiguous -> compiler emits 2x global_load_b128.
__device__ __forceinline__ v16bf ldA(const unsigned short* base, int lda, int k0, int lane) {
  int m = lane & 15, hi = lane >> 4;
  const unsigned short* row = base + m * lda + k0;
  V16U t;
#pragma unroll
  for (int j = 0; j < 8; ++j) {
    int kb = ((j >= 4) ? 16 : 0) + hi * 8 + (j & 3) * 2;
    t.u[j] = *(const unsigned int*)(row + kb);
  }
  return t.v;
}

// B tile 32x16 bf16 where B[k][n] = W[n][k], W row-major [N, ldb].
// Lane = n (+ K half); same K layout as ldA: half h holds K 8h..8h+7 and 16+8h..16+8h+7.
__device__ __forceinline__ v16bf ldB(const unsigned short* W, int ldb, int n0, int k0,
                                     int N, int lane) {
  int n = n0 + (lane & 15), khi = lane >> 4;
  V16U t;
  const unsigned short* row = W + (size_t)n * ldb + k0 + khi * 8;
#pragma unroll
  for (int j = 0; j < 8; ++j)
    t.u[j] = (n < N) ? *(const unsigned int*)(row + ((j >= 4) ? 16 : 0) + (j & 3) * 2) : 0u;
  return t.v;
}

__device__ __forceinline__ v8f wmma_bf16(v16bf a, v16bf b, v8f c) {
  return __builtin_amdgcn_wmma_f32_16x16x32_bf16(false, a, false, b, (short)0, c, false, false);
}

__device__ __forceinline__ float sigm(float x) { return 1.0f / (1.0f + expf(-x)); }

// ---------------- prep kernels ----------------
__global__ void k_zero16(unsigned short* p, int n) {
  int i = blockIdx.x * 256 + threadIdx.x;
  if (i < n) p[i] = 0;
}

// f32 [rows, scols] -> bf16 [rows, dcols], zero-padded columns
__global__ void k_cvt(const float* __restrict__ s, unsigned short* __restrict__ d,
                      int rows, int scols, int dcols) {
  long long i = (long long)blockIdx.x * 256 + threadIdx.x;
  if (i >= (long long)rows * dcols) return;
  int r = (int)(i / dcols), c = (int)(i % dcols);
  d[i] = f2bf(c < scols ? s[(long long)r * scols + c] : 0.0f);
}

// dst[t*128+h] = bf16(W[x[t]*128+h])   (t over B*L)
__global__ void k_gather(const int* __restrict__ x, const float* __restrict__ W,
                         unsigned short* __restrict__ d, int nTok) {
  long long i = (long long)blockIdx.x * 256 + threadIdx.x;
  if (i >= (long long)nTok * 128) return;
  int t = (int)(i >> 7), h = (int)(i & 127);
  d[i] = f2bf(W[(long long)x[t] * 128 + h]);
}

// ---------------- conv-as-GEMM: wave = 16 batch x 64 channels (4 N tiles, shared A) ----
__global__ void k_conv(const unsigned short* __restrict__ e, const unsigned short* __restrict__ W,
                       const float* __restrict__ bias, unsigned short* __restrict__ flat,
                       int K, int Lo, int segOff) {
  int wave = (blockIdx.x * blockDim.x + threadIdx.x) >> 5;
  int lane = threadIdx.x & 31;
  int total = 128 * Lo * 2;            // (B/16) * Lo * (128/64)
  if (wave >= total) return;
  int g = wave & 1, rem = wave >> 1;
  int l = rem % Lo, bt = rem / Lo;
  const unsigned short* abase = e + (size_t)bt * 16 * 6400 + l * 128;  // window is contiguous
  v8f acc[4] = {};
  for (int k0 = 0; k0 < K; k0 += 32) {
    v16bf a = ldA(abase, 6400, k0, lane);
#pragma unroll
    for (int j = 0; j < 4; ++j)
      acc[j] = wmma_bf16(a, ldB(W, K, (g * 4 + j) * 16, k0, 128, lane), acc[j]);
  }
  int hi = lane >> 4;
#pragma unroll
  for (int j = 0; j < 4; ++j) {
    int n = (g * 4 + j) * 16 + (lane & 15);
    float bb = bias[n];
#pragma unroll
    for (int r = 0; r < 8; ++r) {
      int brow = bt * 16 + r + hi * 8;
      float y = fmaxf(acc[j][r] + bb, 0.0f);
      flat[(long long)brow * 23808 + segOff + (long long)n * Lo + l] = f2bf(y);
    }
  }
}

// ------- fc: [2048,23808] @ W^T -> h bf16 [2048,480]; wave = 16 x 64 (4 N tiles) -------
__global__ void k_fc(const unsigned short* __restrict__ flat, const unsigned short* __restrict__ W,
                     const float* __restrict__ bias, unsigned short* __restrict__ hbf) {
  int wave = (blockIdx.x * blockDim.x + threadIdx.x) >> 5;
  int lane = threadIdx.x & 31;
  const int NG = 8;                    // 8 groups of 4 tiles cover 29 N-tiles (450 cols)
  int total = 128 * NG;
  if (wave >= total) return;
  int g = wave % NG, bt = wave / NG;
  const unsigned short* abase = flat + (size_t)bt * 16 * 23808;
  v8f acc[4] = {};
  for (int k0 = 0; k0 < 23808; k0 += 32) {
    v16bf a = ldA(abase, 23808, k0, lane);
#pragma unroll
    for (int j = 0; j < 4; ++j)
      acc[j] = wmma_bf16(a, ldB(W, 23808, (g * 4 + j) * 16, k0, 450, lane), acc[j]);
  }
  int hi = lane >> 4;
#pragma unroll
  for (int j = 0; j < 4; ++j) {
    int n = (g * 4 + j) * 16 + (lane & 15);
    if (n < 450) {
      float bb = bias[n];
#pragma unroll
      for (int r = 0; r < 8; ++r) {
        float y = fmaxf(acc[j][r] + bb, 0.0f);
        hbf[(size_t)(bt * 16 + r + hi * 8) * 480 + n] = f2bf(y);
      }
    }
  }
}

// ------- z heads fused: one wave computes mean AND std tile (shared A) -------
__global__ void k_zheads(const unsigned short* __restrict__ hbf,
                         const unsigned short* __restrict__ Wm, const unsigned short* __restrict__ Ws,
                         const float* __restrict__ bm, const float* __restrict__ bs,
                         float* __restrict__ zm, float* __restrict__ zs) {
  int wave = (blockIdx.x * blockDim.x + threadIdx.x) >> 5;
  int lane = threadIdx.x & 31;
  const int NT = 19, total = 128 * NT;
  if (wave >= total) return;
  int nt = wave % NT, bt = wave / NT;
  const unsigned short* abase = hbf + (size_t)bt * 16 * 480;
  v8f am = {}, as = {};
  for (int k0 = 0; k0 < 480; k0 += 32) {
    v16bf a = ldA(abase, 480, k0, lane);
    am = wmma_bf16(a, ldB(Wm, 480, nt * 16, k0, 300, lane), am);
    as = wmma_bf16(a, ldB(Ws, 480, nt * 16, k0, 300, lane), as);
  }
  int n = nt * 16 + (lane & 15), hi = lane >> 4;
  if (n < 300) {
#pragma unroll
    for (int r = 0; r < 8; ++r) {
      size_t idx = (size_t)(bt * 16 + r + hi * 8) * 300 + n;
      zm[idx] = am[r] + bm[n];
      zs[idx] = as[r] + bs[n];
    }
  }
}

// z = zm + zs*300  -> bf16 [2048,320pad]
__global__ void k_zcombine(const float* __restrict__ zm, const float* __restrict__ zs,
                           unsigned short* __restrict__ zbf) {
  int i = blockIdx.x * 256 + threadIdx.x;
  if (i >= 2048 * 300) return;
  int b = i / 300, n = i % 300;
  zbf[(size_t)b * 320 + n] = f2bf(zm[i] + zs[i] * 300.0f);
}

// per-row KL, deterministic
__global__ void k_latent_row(const float* __restrict__ zm, const float* __restrict__ zs,
                             float* __restrict__ lat) {
  int b = blockIdx.x * 256 + threadIdx.x;
  if (b >= 2048) return;
  float s = 0.0f;
  for (int n = 0; n < 300; ++n) {
    float m = zm[(size_t)b * 300 + n], v = zs[(size_t)b * 300 + n];
    s += m * m + v * v - logf(v * v) - 1.0f;
  }
  lat[b] = 0.5f * s;
}

__global__ void k_latent_reduce(const float* __restrict__ lat, float* __restrict__ out) {
  __shared__ float sm[256];
  int tid = threadIdx.x;
  float s = 0.0f;
  for (int i = tid; i < 2048; i += 256) s += lat[i];
  sm[tid] = s;
  __syncthreads();
  for (int st = 128; st > 0; st >>= 1) {
    if (tid < st) sm[tid] += sm[tid + st];
    __syncthreads();
  }
  if (tid == 0) out[0] = sm[0] / 2048.0f;
}

// ---------------- decoder init: zd = relu(z @ dec_W^T + b) ----------------
__global__ void k_dec(const unsigned short* __restrict__ zbf, const unsigned short* __restrict__ W,
                      const float* __restrict__ bias, unsigned short* __restrict__ zd_bf,
                      float* __restrict__ zd_f) {
  int wave = (blockIdx.x * blockDim.x + threadIdx.x) >> 5;
  int lane = threadIdx.x & 31;
  const int NT = 10, total = 128 * NT;
  if (wave >= total) return;
  int nt = wave % NT, bt = wave / NT;
  const unsigned short* abase = zbf + (size_t)bt * 16 * 320;
  v8f acc = {};
  for (int k0 = 0; k0 < 320; k0 += 32) {
    v16bf a = ldA(abase, 320, k0, lane);
    acc = wmma_bf16(a, ldB(W, 320, nt * 16, k0, 150, lane), acc);
  }
  int n = nt * 16 + (lane & 15), hi = lane >> 4;
  if (n < 150) {
    float bb = bias[n];
#pragma unroll
    for (int r = 0; r < 8; ++r) {
      int brow = bt * 16 + r + hi * 8;
      float y = fmaxf(acc[r] + bb, 0.0f);
      zd_f[(size_t)brow * 150 + n] = y;
      zd_bf[(size_t)brow * 160 + n] = f2bf(y);
    }
  }
}

// ---------------- recurrent decoder: one block owns 16 batch rows for all 50 steps ------
// Gates GEMM: each wave owns 5 consecutive N-tiles with the A tile hoisted/shared.
__global__ void __launch_bounds__(256)
k_lstm(const unsigned short* __restrict__ de, const unsigned short* __restrict__ zd_bf,
       const float* __restrict__ zd_f, const unsigned short* __restrict__ Wih,
       const unsigned short* __restrict__ Whh, const float* __restrict__ bih,
       const float* __restrict__ bhh, const unsigned short* __restrict__ Wout,
       const float* __restrict__ outB, float* __restrict__ out) {
  __shared__ float          gates[16][608];
  __shared__ float          cf[16][160];
  __shared__ unsigned short cbf[16][160];
  __shared__ unsigned short hbf[16][160];
  int tid = threadIdx.x, lane = tid & 31, w = tid >> 5;
  int b0 = blockIdx.x * 16;

  for (int i = tid; i < 16 * 160; i += 256) {
    int r = i / 160, c = i % 160;
    cbf[r][c] = zd_bf[(size_t)(b0 + r) * 160 + c];     // pads already zero
    cf[r][c]  = (c < 150) ? zd_f[(size_t)(b0 + r) * 150 + c] : 0.0f;
    hbf[r][c] = 0;
  }
  __syncthreads();

  for (int t = 0; t < 50; ++t) {
    // gates = x_t @ W_ih^T + c @ W_hh^T + b_ih + b_hh   (N = 600 -> 38 tiles, 5 per wave;
    // tiles 38,39 are dead: ldB's n<600 guard feeds zeros, epilogue skips them)
    const unsigned short* xbase = de + (size_t)b0 * 6400 + t * 128;
    {
      int base = w * 5;
      v8f acc[5] = {};
      for (int k0 = 0; k0 < 128; k0 += 32) {
        v16bf a = ldA(xbase, 6400, k0, lane);
#pragma unroll
        for (int j = 0; j < 5; ++j)
          acc[j] = wmma_bf16(a, ldB(Wih, 128, (base + j) * 16, k0, 600, lane), acc[j]);
      }
      for (int k0 = 0; k0 < 160; k0 += 32) {
        v16bf a = ldA(&cbf[0][0], 160, k0, lane);
#pragma unroll
        for (int j = 0; j < 5; ++j)
          acc[j] = wmma_bf16(a, ldB(Whh, 160, (base + j) * 16, k0, 600, lane), acc[j]);
      }
      int hi = lane >> 4;
#pragma unroll
      for (int j = 0; j < 5; ++j) {
        int n = (base + j) * 16 + (lane & 15);
        if (n < 600) {
          float bb = bih[n] + bhh[n];
#pragma unroll
          for (int r = 0; r < 8; ++r) gates[r + hi * 8][n] = acc[j][r] + bb;
        }
      }
    }
    __syncthreads();

    // elementwise cell update (reference passes c as both h and c)
    for (int i = tid; i < 16 * 150; i += 256) {
      int r = i / 150, c = i % 150;
      float ig = gates[r][c], fg = gates[r][150 + c];
      float gg = gates[r][300 + c], og = gates[r][450 + c];
      float cp = cf[r][c];
      float cn = sigm(fg) * cp + sigm(ig) * tanhf(gg);
      float hv = sigm(og) * tanhf(cn);
      cf[r][c] = cn;
      cbf[r][c] = f2bf(cn);
      hbf[r][c] = f2bf(hv);
    }
    __syncthreads();

    // logits_t = h @ out_W^T + out_b   (N = 56 -> 4 tiles)
    if (w < 4) {
      int n0 = w * 16;
      v8f acc = {};
      for (int k0 = 0; k0 < 160; k0 += 32) {
        v16bf a = ldA(&hbf[0][0], 160, k0, lane);
        acc = wmma_bf16(a, ldB(Wout, 160, n0, k0, 56, lane), acc);
      }
      int n = n0 + (lane & 15), hi = lane >> 4;
      if (n < 56) {
        float bb = outB[n];
#pragma unroll
        for (int r = 0; r < 8; ++r) {
          int brow = b0 + r + hi * 8;
          out[1 + ((long long)brow * 50 + t) * 56 + n] = acc[r] + bb;
        }
      }
    }
    __syncthreads();
  }
}

// ---------------- launcher ----------------
extern "C" void kernel_launch(void* const* d_in, const int* in_sizes, int n_in,
                              void* d_out, int out_size, void* d_ws, size_t ws_size,
                              hipStream_t stream) {
  (void)in_sizes; (void)n_in; (void)out_size; (void)ws_size;
  const int*   x      = (const int*)  d_in[0];
  const float* embedW = (const float*)d_in[1];
  const float* dembW  = (const float*)d_in[2];
  const float* fcW    = (const float*)d_in[3];
  const float* fcB    = (const float*)d_in[4];
  const float* fmW    = (const float*)d_in[5];
  const float* fmB    = (const float*)d_in[6];
  const float* fsW    = (const float*)d_in[7];
  const float* fsB    = (const float*)d_in[8];
  const float* decW   = (const float*)d_in[9];
  const float* decB   = (const float*)d_in[10];
  const float* Wih    = (const float*)d_in[11];
  const float* Whh    = (const float*)d_in[12];
  const float* bih    = (const float*)d_in[13];
  const float* bhh    = (const float*)d_in[14];
  const float* outW   = (const float*)d_in[15];
  const float* outB   = (const float*)d_in[16];
  const float* convW[4] = {(const float*)d_in[17], (const float*)d_in[19],
                           (const float*)d_in[21], (const float*)d_in[23]};
  const float* convB[4] = {(const float*)d_in[18], (const float*)d_in[20],
                           (const float*)d_in[22], (const float*)d_in[24]};
  float* out = (float*)d_out;

  char* ws = (char*)d_ws;
  size_t off = 0;
  auto alloc = [&](size_t bytes) -> char* {
    char* p = ws + off;
    off = (off + bytes + 255) & ~(size_t)255;
    return p;
  };
  unsigned short* e_bf    = (unsigned short*)alloc(2048ll * 50 * 128 * 2);
  unsigned short* de_bf   = (unsigned short*)alloc(2048ll * 50 * 128 * 2);
  unsigned short* flat_bf = (unsigned short*)alloc(2048ll * 23808 * 2);
  unsigned short* fcW_bf  = (unsigned short*)alloc(450ll * 23808 * 2);
  unsigned short* cW_bf[4];
  const int KK[4] = {3, 4, 5, 6}, LO[4] = {48, 47, 46, 45}, SEG[4] = {0, 6144, 12160, 18048};
  for (int j = 0; j < 4; ++j) cW_bf[j] = (unsigned short*)alloc((size_t)128 * KK[j] * 128 * 2);
  unsigned short* fmW_bf  = (unsigned short*)alloc(300ll * 480 * 2);
  unsigned short* fsW_bf  = (unsigned short*)alloc(300ll * 480 * 2);
  unsigned short* decW_bf = (unsigned short*)alloc(150ll * 320 * 2);
  unsigned short* Wih_bf  = (unsigned short*)alloc(600ll * 128 * 2);
  unsigned short* Whh_bf  = (unsigned short*)alloc(600ll * 160 * 2);
  unsigned short* outW_bf = (unsigned short*)alloc(56ll * 160 * 2);
  unsigned short* h_bf    = (unsigned short*)alloc(2048ll * 480 * 2);
  float*          zm_f    = (float*)alloc(2048ll * 300 * 4);
  float*          zs_f    = (float*)alloc(2048ll * 300 * 4);
  unsigned short* z_bf    = (unsigned short*)alloc(2048ll * 320 * 2);
  unsigned short* zd_bf   = (unsigned short*)alloc(2048ll * 160 * 2);
  float*          zd_f    = (float*)alloc(2048ll * 150 * 4);
  float*          lat     = (float*)alloc(2048ll * 4);

  auto gb = [](long long n) { return (int)((n + 255) / 256); };  // elementwise grids
  auto wb = [](int waves)   { return (waves + 7) / 8; };         // wave-per-tile grids

  // zero-fill padded activation buffers (poison-safe, every call)
  k_zero16<<<gb(2048ll * 480), 256, 0, stream>>>(h_bf, 2048 * 480);
  k_zero16<<<gb(2048ll * 320), 256, 0, stream>>>(z_bf, 2048 * 320);
  k_zero16<<<gb(2048ll * 160), 256, 0, stream>>>(zd_bf, 2048 * 160);

  // weight conversions (with K zero-padding)
  k_cvt<<<gb(450ll * 23808), 256, 0, stream>>>(fcW, fcW_bf, 450, 23808, 23808);
  for (int j = 0; j < 4; ++j)
    k_cvt<<<gb(128ll * KK[j] * 128), 256, 0, stream>>>(convW[j], cW_bf[j], 128, KK[j] * 128, KK[j] * 128);
  k_cvt<<<gb(300ll * 480), 256, 0, stream>>>(fmW, fmW_bf, 300, 450, 480);
  k_cvt<<<gb(300ll * 480), 256, 0, stream>>>(fsW, fsW_bf, 300, 450, 480);
  k_cvt<<<gb(150ll * 320), 256, 0, stream>>>(decW, decW_bf, 150, 300, 320);
  k_cvt<<<gb(600ll * 128), 256, 0, stream>>>(Wih, Wih_bf, 600, 128, 128);
  k_cvt<<<gb(600ll * 160), 256, 0, stream>>>(Whh, Whh_bf, 600, 150, 160);
  k_cvt<<<gb(56ll * 160), 256, 0, stream>>>(outW, outW_bf, 56, 150, 160);

  // embedding gathers
  k_gather<<<gb(2048ll * 50 * 128), 256, 0, stream>>>(x, embedW, e_bf, 2048 * 50);
  k_gather<<<gb(2048ll * 50 * 128), 256, 0, stream>>>(x, dembW, de_bf, 2048 * 50);

  // convs (A rows are contiguous k*128 windows of e); wave = 16 batch x 64 channels
  for (int j = 0; j < 4; ++j)
    k_conv<<<wb(128 * LO[j] * 2), 256, 0, stream>>>(e_bf, cW_bf[j], convB[j], flat_bf,
                                                    KK[j] * 128, LO[j], SEG[j]);

  // big FC: wave = 16 x 64
  k_fc<<<wb(128 * 8), 256, 0, stream>>>(flat_bf, fcW_bf, fcB, h_bf);

  // z heads (fused mean+std) + latent loss + z
  k_zheads<<<wb(128 * 19), 256, 0, stream>>>(h_bf, fmW_bf, fsW_bf, fmB, fsB, zm_f, zs_f);
  k_zcombine<<<gb(2048ll * 300), 256, 0, stream>>>(zm_f, zs_f, z_bf);
  k_latent_row<<<gb(2048), 256, 0, stream>>>(zm_f, zs_f, lat);
  k_latent_reduce<<<1, 256, 0, stream>>>(lat, out);

  // decoder init state
  k_dec<<<wb(128 * 10), 256, 0, stream>>>(z_bf, decW_bf, decB, zd_bf, zd_f);

  // recurrent decoder + output projection
  k_lstm<<<128, 256, 0, stream>>>(de_bf, zd_bf, zd_f, Wih_bf, Whh_bf, bih, bhh,
                                  outW_bf, outB, out);
}
